// get_lr_40338332844775
// MI455X (gfx1250) — compile-verified, matrix operand order checked by reading, not run on hardware
//
#include <hip/hip_runtime.h>

// MI455X / gfx1250, wave32, WMMA + TDM.
// Pipeline: pool (134MB NT streaming read) -> small f32 WMMA GEMM + sigmoid
//        -> f16-input / f32-accum WMMA reconstruction GEMM (134MB NT write).
// Roofline: ~268MB @ 23.3 TB/s => ~11.5us floor; f16 WMMA keeps recon compute
// (~4.3 GFLOP, 2x 16x16x32 per tile) ~2us. LDS staging uses TENSOR_LOAD_TO_LDS.

typedef _Float16 h2   __attribute__((ext_vector_type(2)));
typedef _Float16 v16h __attribute__((ext_vector_type(16)));
typedef float    v2f  __attribute__((ext_vector_type(2)));
typedef float    v4f  __attribute__((ext_vector_type(4)));
typedef float    v8f  __attribute__((ext_vector_type(8)));

#define BD   128        // batch
#define DD   64         // c == h == w == 64
#define VOL  (DD*DD*DD) // 262144 elems per batch

// ---------------- TDM helper: 1-D contiguous global -> LDS copy ----------
// D# per CDNA5 ISA ch.8: group0 = {flags, lds_addr, global_addr, type=2},
// group1 = {data_size=8B, tensor_dim0=tile_dim0=n8, tensor_dim1=tile_dim1=1,
// tensor_dim0_stride=n8}. Issued by one wave; completion via TENSORcnt.
// This toolchain exposes the 6-arg builtin:
//   (uint32x4 g0, int32x8 g1, int32x4 g2, int32x4 g3, int32x8 extra, i32 cpol)
#if __has_builtin(__builtin_amdgcn_tensor_load_to_lds) && \
    __has_builtin(__builtin_amdgcn_s_wait_tensorcnt)
#define HAVE_TDM 1
typedef unsigned int v4u  __attribute__((ext_vector_type(4)));
typedef int          v8i_ __attribute__((ext_vector_type(8)));
typedef int          v4i_ __attribute__((ext_vector_type(4)));

__device__ __forceinline__ void tdm_copy_1d(void* lds_dst, const void* gsrc,
                                            unsigned n8 /* 8-byte units */) {
  unsigned long long ga = (unsigned long long)(size_t)gsrc;
  unsigned la = (unsigned)(size_t)lds_dst;  // flat->LDS: addr[31:0] is LDS byte addr
  v4u g0 = { 1u,                                  // count=1, user-mode
             la,                                  // lds_addr [63:32]
             (unsigned)ga,                        // global_addr [95:64]
             ((unsigned)(ga >> 32) & 0x01FFFFFFu) // global_addr [120:96]
               | (2u << 30) };                    // type=2 ("image")
  v8i_ g1 = { (int)(3u << 16),                    // data_size = 8 bytes
              (int)((n8 & 0xFFFFu) << 16),        // tensor_dim0 lo16
              (int)((n8 >> 16) | (1u << 16)),     // tensor_dim0 hi16, tensor_dim1=1
              (int)((n8 & 0xFFFFu) << 16),        // tile_dim0 = n8
              1,                                  // tile_dim1 = 1
              (int)n8, 0, 0 };                    // tensor_dim0_stride = n8
  v4i_ z4 = { 0, 0, 0, 0 };
  v8i_ z8 = { 0, 0, 0, 0, 0, 0, 0, 0 };
  __builtin_amdgcn_tensor_load_to_lds(g0, g1, z4, z4, z8, 0);
}
#endif

// ---------------- Kernel 0: zero pooled-sum accumulators ----------------
__global__ void zero_vecs_kernel(float* __restrict__ v, int n) {
  int i = blockIdx.x * blockDim.x + threadIdx.x;
  if (i < n) v[i] = 0.0f;
}

// ---------------- Kernel 1: triple adaptive-avg-pool (sums) -------------
// c_vec[b,c] = sum over (h,w); h_vec[b,w] = sum over (c,h);
// w_vec[b,h] = sum over (c,w).  1/4096 folded into kernel 2's sigmoid arg.
// b128 NT loads: each wave covers 2 rows (128 floats) per load; row sums by
// half-wave shfl reduction; per-lane w-partials accumulate in registers.
__global__ void pool3_kernel(const float* __restrict__ x,
                             float* __restrict__ cvec,
                             float* __restrict__ hvec,
                             float* __restrict__ wvec) {
  __shared__ float s_c[DD], s_h[DD], s_w[DD];
  const int b   = blockIdx.x;
  const int tid = threadIdx.x;
  if (tid < DD) { s_c[tid] = 0.f; s_h[tid] = 0.f; s_w[tid] = 0.f; }
  __syncthreads();

  const int wave = tid >> 5, lane = tid & 31;
  const int half = lane >> 4;       // which of the 2 rows in this load
  const int l    = lane & 15;       // covers w = 4l .. 4l+3
  const float* base = x + (size_t)b * VOL;
  const int pr0 = blockIdx.y * 512; // row-pairs chunk (4 chunks per batch)
  float ha0 = 0.f, ha1 = 0.f, ha2 = 0.f, ha3 = 0.f;
  for (int pr = pr0 + wave; pr < pr0 + 512; pr += 8) {
    const int r = pr * 2 + half;    // this half-wave's row; rows 2pr,2pr+1 share c
    const int c = r >> 6, h = r & 63;
    v4f v = __builtin_nontemporal_load(
        (const v4f*)(base + (size_t)r * 64 + l * 4));
    ha0 += v[0]; ha1 += v[1]; ha2 += v[2]; ha3 += v[3];
    float rs = (v[0] + v[1]) + (v[2] + v[3]);
    for (int m = 8; m >= 1; m >>= 1) rs += __shfl_xor(rs, m, 32); // 16-lane half
    if (l == 0) { atomicAdd(&s_c[c], rs); atomicAdd(&s_w[h], rs); }
  }
  atomicAdd(&s_h[4 * l + 0], ha0);
  atomicAdd(&s_h[4 * l + 1], ha1);
  atomicAdd(&s_h[4 * l + 2], ha2);
  atomicAdd(&s_h[4 * l + 3], ha3);
  __syncthreads();
  if (tid < DD) {
    atomicAdd(&cvec[b * DD + tid], s_c[tid]);
    atomicAdd(&hvec[b * DD + tid], s_h[tid]);
    atomicAdd(&wvec[b * DD + tid], s_w[tid]);
  }
}

// ---------------- Kernel 2: [128x64] x [64x4096] f32 WMMA + sigmoid -----
// out[b, i*64+o] = sigmoid( (1/4096) * sum_j W[i,o,j] * vecsum[b,j] ),
// stored transposed as f16 T[branch][b][o][i].
__global__ void small_gemm_sigmoid_kernel(const float* __restrict__ vecs, // [3][B][64] sums
                                          const float* __restrict__ Wc,
                                          const float* __restrict__ Wh,
                                          const float* __restrict__ Ww,
                                          _Float16* __restrict__ Tout) {  // [3][B][64][64]
  __shared__ float s_vec[BD * DD]; // 32 KB: all 128 descriptor rows
  __shared__ float s_w[64 * DD];   // 16 KB: 64-col chunk of W^T
  const int branch = blockIdx.y;
  const int n0     = blockIdx.x * 64;  // io chunk
  const int tid    = threadIdx.x;
  const float* W   = (branch == 0) ? Wc : (branch == 1) ? Wh : Ww;
  const float* vec = vecs + (size_t)branch * (BD * DD);
#ifdef HAVE_TDM
  if (tid < 32) {                     // wave 0 drives the DMA engine
    tdm_copy_1d(s_vec, vec, (BD * DD) / 2);
    tdm_copy_1d(s_w, W + (size_t)n0 * 64, (64 * DD) / 2);
    __builtin_amdgcn_s_wait_tensorcnt(0);
  }
#else
  for (int k = tid; k < BD * DD; k += 256) s_vec[k] = vec[k];
  for (int k = tid; k < 64 * DD; k += 256) s_w[k] = W[(size_t)n0 * 64 + k];
#endif
  __syncthreads();

  const int wave = tid >> 5, lane = tid & 31;
  const int g = lane >> 4, nl = lane & 15;     // wmma lane decomposition
  const int arow = wave * 16 + nl;             // A-matrix M (batch row)
  const float scale = 1.0f / 4096.0f;

  for (int nt = 0; nt < 4; ++nt) {
    const int col = nt * 16 + nl;              // B-matrix N (local io col)
    v8f acc = {};
#pragma unroll
    for (int j0 = 0; j0 < 64; j0 += 4) {
      // 16x4 f32 A/B layout: per lane K = v + 2*g  (v = VGPR index)
      v2f a  = *(const v2f*)&s_vec[arow * DD + j0 + 2 * g];
      v2f bb = *(const v2f*)&s_w [col  * DD + j0 + 2 * g];
      acc = __builtin_amdgcn_wmma_f32_16x16x4_f32(false, a, false, bb,
                                                  (short)0, acc, false, false);
    }
#pragma unroll
    for (int r = 0; r < 8; ++r) {              // C layout: M = 8*g + r
      const int bIdx = wave * 16 + g * 8 + r;
      const int io   = n0 + nt * 16 + nl;
      const int i    = io >> 6, o = io & 63;
      const float z  = acc[r] * scale;
      const float s  = __builtin_amdgcn_rcpf(1.0f + __expf(-z)); // v_rcp_f32
      Tout[(size_t)branch * (BD * DD * DD) +
           ((size_t)bIdx * DD + o) * DD + i] = (_Float16)s;
    }
  }
}

// ---------------- Kernel 3: reconstruction GEMM (f16 WMMA, f32 accum) ---
// out[b,c,p,q] = sum_i fc[b,c,i] * fh[b,p,i] * fw[b,q,i]
// Per block: one b, two p values. A = fc^T tile; B[i,q] = fh[p,i]*fw[q,i]
// built in-register (v_pk_mul_f16) directly in the WMMA layout.
__global__ void recon_kernel(const _Float16* __restrict__ T,
                             float* __restrict__ out) {
  __shared__ _Float16 s_fc[DD * DD];  // [c][i]  8 KB
  __shared__ _Float16 s_fw[DD * DD];  // [q][i]  8 KB
  __shared__ _Float16 s_fh[2 * DD];   // [pp][i] 256 B
  const int b   = blockIdx.x;
  const int p0  = blockIdx.y * 2;
  const int tid = threadIdx.x;
  const _Float16* gfc = T + (size_t)0 * (BD * DD * DD) + (size_t)b * DD * DD;
  const _Float16* gfh = T + (size_t)1 * (BD * DD * DD) + (size_t)b * DD * DD;
  const _Float16* gfw = T + (size_t)2 * (BD * DD * DD) + (size_t)b * DD * DD;

#ifdef HAVE_TDM
  if (tid < 32) {                     // wave 0 drives the DMA engine
    tdm_copy_1d(s_fc, gfc, (DD * DD) / 4);                    // 8 KB
    tdm_copy_1d(s_fw, gfw, (DD * DD) / 4);                    // 8 KB
    tdm_copy_1d(s_fh, gfh + (size_t)p0 * DD, (2 * DD) / 4);   // 256 B
    __builtin_amdgcn_s_wait_tensorcnt(0);
  }
#else
  {
    const uint32_t* c32 = (const uint32_t*)gfc;
    const uint32_t* w32 = (const uint32_t*)gfw;
    uint32_t* sc32 = (uint32_t*)s_fc;
    uint32_t* sw32 = (uint32_t*)s_fw;
    for (int k = tid; k < DD * DD / 2; k += 256) { sc32[k] = c32[k]; sw32[k] = w32[k]; }
    if (tid < 64) {
      const int pp = tid >> 5, off = tid & 31;
      ((uint32_t*)s_fh)[pp * 32 + off] =
          ((const uint32_t*)(gfh + (size_t)(p0 + pp) * DD))[off];
    }
  }
#endif
  __syncthreads();

  const int wave = tid >> 5, lane = tid & 31;
  const int g = lane >> 4, nl = lane & 15;
  const int ploc = wave >> 2, qt = wave & 3;   // 2 p x 4 q-tiles over 8 waves
  const int q = qt * 16 + nl;
  const int p = p0 + ploc;

  // B fragments for K = 0..31 and K = 32..63 (shared across the 4 m-tiles).
  // 16-bit layout: per lane, VGPR v holds K = 16*(v/4) + 8*g + 2*(v%4) + {0,1}
  v16h bf0, bf1;
#pragma unroll
  for (int v = 0; v < 8; ++v) {
    const int Kb = ((v >> 2) << 4) + (g << 3) + ((v & 3) << 1);
    h2 f0 = *(const h2*)&s_fh[ploc * DD + Kb];
    h2 f1 = *(const h2*)&s_fh[ploc * DD + 32 + Kb];
    h2 w0 = *(const h2*)&s_fw[q * DD + Kb];
    h2 w1 = *(const h2*)&s_fw[q * DD + 32 + Kb];
    h2 r0 = f0 * w0;                      // v_pk_mul_f16
    h2 r1 = f1 * w1;
    bf0[2 * v] = r0[0]; bf0[2 * v + 1] = r0[1];
    bf1[2 * v] = r1[0]; bf1[2 * v + 1] = r1[1];
  }

  for (int m = 0; m < 4; ++m) {
    const int c = m * 16 + nl;
    v16h a0, a1;
#pragma unroll
    for (int v = 0; v < 8; ++v) {
      const int Kb = ((v >> 2) << 4) + (g << 3) + ((v & 3) << 1);
      h2 t0 = *(const h2*)&s_fc[c * DD + Kb];
      h2 t1 = *(const h2*)&s_fc[c * DD + 32 + Kb];
      a0[2 * v] = t0[0]; a0[2 * v + 1] = t0[1];
      a1[2 * v] = t1[0]; a1[2 * v + 1] = t1[1];
    }
    v8f acc = {};
    acc = __builtin_amdgcn_wmma_f32_16x16x32_f16(false, a0, false, bf0,
                                                 (short)0, acc, false, false);
    acc = __builtin_amdgcn_wmma_f32_16x16x32_f16(false, a1, false, bf1,
                                                 (short)0, acc, false, false);
#pragma unroll
    for (int r = 0; r < 8; ++r) {
      const int cc = m * 16 + g * 8 + r;
      __builtin_nontemporal_store(
          acc[r], &out[(((size_t)b * DD + cc) * DD + p) * DD + q]);
    }
  }
}

// ---------------- Host-side launcher ------------------------------------
extern "C" void kernel_launch(void* const* d_in, const int* in_sizes, int n_in,
                              void* d_out, int out_size, void* d_ws, size_t ws_size,
                              hipStream_t stream) {
  const float* x  = (const float*)d_in[0];
  const float* Wc = (const float*)d_in[1];
  const float* Wh = (const float*)d_in[2];
  const float* Ww = (const float*)d_in[3];

  // Workspace: [3][B][64] f32 pooled sums (96 KB), then [3][B][64][64] f16 (3 MB)
  float* vecs = (float*)d_ws;
  _Float16* T = (_Float16*)((char*)d_ws + (size_t)3 * BD * DD * sizeof(float));

  const int nvec = 3 * BD * DD;
  zero_vecs_kernel<<<(nvec + 255) / 256, 256, 0, stream>>>(vecs, nvec);
  pool3_kernel<<<dim3(BD, 4), 256, 0, stream>>>(x, vecs, vecs + BD * DD,
                                                vecs + 2 * BD * DD);
  small_gemm_sigmoid_kernel<<<dim3(64, 3), 256, 0, stream>>>(vecs, Wc, Wh, Ww, T);
  recon_kernel<<<dim3(BD, 32), 256, 0, stream>>>(T, (float*)d_out);
}
